// LinearQuant_51333449122264
// MI455X (gfx1250) — matrix operand order repaired and checked by reading, not compiled, not run on hardware
//
#include <hip/hip_runtime.h>
#include <stdint.h>

// LinearQuant: out[8192,16] = (int8 W * 2^-7) @ x[8192,16]
// Weight-bandwidth-bound (64MB int8 stream, AI ~33 FLOP/B << WMMA roofline).
// Stream W once; expand int8->f16 EXACTLY with the 0x6400 magic-number trick
// (xor + v_perm_b32 + v_pk_add_f16: 5 VALU / 4 bytes); f16 WMMA w/ f32 acc;
// split-K x8 per block with LDS reduction; x pre-transposed to f16 in d_ws;
// explicit 1-deep software pipeline so next step's loads issue before the WMMA.

typedef __attribute__((ext_vector_type(16))) _Float16 v16h;
typedef __attribute__((ext_vector_type(8)))  float    v8f;
typedef __attribute__((ext_vector_type(8)))  uint32_t v8u;
typedef __attribute__((ext_vector_type(2)))  _Float16 h2;

constexpr int OUT_F = 8192;
constexpr int IN_F  = 8192;
constexpr int NCOL  = 16;
constexpr int WAVES = 8;                   // waves per block (256 threads)
constexpr int KSPLIT = IN_F / WAVES;       // 1024 K per wave
constexpr int KSTEP  = 32;                 // K per WMMA_F32_16X16X32_F16
constexpr int NSTEP  = KSPLIT / KSTEP;     // 32 WMMA steps per wave

// ---- pre-pass: x[k][n] (f32, row-major) -> xT[n][k] (f16) so B loads are contiguous
__global__ __launch_bounds__(256) void lq_xcvt_transpose(
    const float* __restrict__ x, _Float16* __restrict__ xT)
{
    const int k = blockIdx.x * 256 + threadIdx.x;   // 0..IN_F-1
    const int n = blockIdx.y;                       // 0..NCOL-1
    xT[(size_t)n * IN_F + k] = (_Float16)x[(size_t)k * NCOL + n];
}

// Two int8 bytes (already xor'ed with 0x80) -> packed f16 pair, exactly.
// as_f16(0x6400 | u) = 1024 + u = b + 1152; subtract 1152 (exact in f16).
__device__ __forceinline__ uint32_t bytes2half2(uint32_t ex, uint32_t sel) {
#if __has_builtin(__builtin_amdgcn_perm)
    uint32_t p = __builtin_amdgcn_perm(0x64646464u, ex, sel);   // v_perm_b32
#else
    uint32_t b0 = (ex >> ((sel & 0xff) * 8)) & 0xff;
    uint32_t b1 = (ex >> (((sel >> 16) & 0xff) * 8)) & 0xff;
    uint32_t p  = 0x64006400u | b0 | (b1 << 16);
#endif
    h2 v = __builtin_bit_cast(h2, p);
    const h2 bias = {(_Float16)(-1152.0f), (_Float16)(-1152.0f)};
    v = v + bias;                                               // v_pk_add_f16
    return __builtin_bit_cast(uint32_t, v);
}

// Raw A fragment: 16 int8 weights for this lane (16 contiguous bytes: K = khalf*16 + {0..15}).
struct AFrag { uint32_t d0, d1, d2, d3; };

__device__ __forceinline__ AFrag load_a(const int8_t* aptr, int s) {
    const uint32_t* ap = (const uint32_t*)(aptr + (size_t)s * KSTEP);
    AFrag f;
    f.d0 = ap[0];  f.d1 = ap[1];   // one global_load_b64
    f.d2 = ap[2];  f.d3 = ap[3];   // one global_load_b64 (offset +8)
    return f;
}

// Expand raw int8 fragment to WMMA A register layout (v16h), exactly.
__device__ __forceinline__ v16h expand_a(AFrag f) {
    const uint32_t e0 = f.d0 ^ 0x80808080u;
    const uint32_t e1 = f.d1 ^ 0x80808080u;
    const uint32_t e2 = f.d2 ^ 0x80808080u;
    const uint32_t e3 = f.d3 ^ 0x80808080u;
    v8u au;
    au[0] = bytes2half2(e0, 0x05010400u);   // K = kb+0,1
    au[1] = bytes2half2(e0, 0x07030602u);   // K = kb+2,3
    au[2] = bytes2half2(e1, 0x05010400u);   // K = kb+4,5
    au[3] = bytes2half2(e1, 0x07030602u);   // K = kb+6,7
    au[4] = bytes2half2(e2, 0x05010400u);   // K = kb+8,9
    au[5] = bytes2half2(e2, 0x07030602u);   // K = kb+10,11
    au[6] = bytes2half2(e3, 0x05010400u);   // K = kb+12,13
    au[7] = bytes2half2(e3, 0x07030602u);   // K = kb+14,15
    return __builtin_bit_cast(v16h, au);
}

// B layout (32x16): lane holds column n=col, K = khalf*16 + {0..15} along VGPRs.
template<bool XF16>
__device__ __forceinline__ v16h load_b(const void* xsrc, int col, int khalf,
                                       int k0, int s) {
    const int kk = k0 + s * KSTEP;
    if (XF16) {
        const _Float16* xT = (const _Float16*)xsrc;   // [NCOL][IN_F]
        return *(const v16h*)(xT + (size_t)col * IN_F + kk + khalf * 16);  // 32B contiguous
    } else {
        const float* xf = (const float*)xsrc;         // [IN_F][NCOL]
        v16h b;
        #pragma unroll
        for (int j = 0; j < 16; ++j)
            b[j] = (_Float16)xf[(size_t)(kk + khalf * 16 + j) * NCOL + col];
        return b;
    }
}

// ---- main GEMM: one 16x16 out tile per block, split-K across 8 waves ----
template<bool XF16>
__global__ __launch_bounds__(256) void lq_wmma_gemm(
    const int8_t* __restrict__ W,
    const void*   __restrict__ xsrc,
    const float*  __restrict__ scaler,
    float*        __restrict__ out)
{
    __shared__ float red[WAVES * 256];

    const int tid   = threadIdx.x;
    const int wv    = tid >> 5;        // wave id 0..7  -> K split
    const int lane  = tid & 31;
    const int mt    = blockIdx.x;      // output row tile 0..511
    const int col   = lane & 15;       // N index (B/C/D lane mapping)
    const int khalf = lane >> 4;       // half-wave selector in K

    const int m  = mt * 16 + col;      // A: lane holds row m = lane&15 of the tile
    const int k0 = wv * KSPLIT;

    // A k order used here (contiguous, same as B): lane holds K = khalf*16 + {0..15}.
    const int8_t* aptr = W + (size_t)m * IN_F + k0 + khalf * 16;

    v8f c = {};

    // ---- software pipeline: loads for step s+1 issue before WMMA of step s ----
    AFrag fa = load_a(aptr, 0);
    v16h  b  = load_b<XF16>(xsrc, col, khalf, k0, 0);

    #pragma unroll 4
    for (int s = 0; s < NSTEP - 1; ++s) {
        AFrag fn = load_a(aptr, s + 1);
        v16h  bn = load_b<XF16>(xsrc, col, khalf, k0, s + 1);
        __builtin_prefetch(aptr + (size_t)s * KSTEP + 512, 0, 1);  // global_prefetch_b8

        const v16h a = expand_a(fa);
        c = __builtin_amdgcn_wmma_f32_16x16x32_f16(
                false, a, false, b, (short)0, c, false, false);

        fa = fn;  b = bn;
    }
    {   // epilogue: last step
        const v16h a = expand_a(fa);
        c = __builtin_amdgcn_wmma_f32_16x16x32_f16(
                false, a, false, b, (short)0, c, false, false);
    }

    // C/D layout: c[i] = C[row = i + 8*khalf][col]. Stash partials in LDS.
    #pragma unroll
    for (int i = 0; i < 8; ++i)
        red[wv * 256 + (i + 8 * khalf) * 16 + col] = c[i];

    __syncthreads();

    // reduce the 8 K-split partials; apply per-tensor scaler; coalesced store
    const float s = scaler[0];
    float acc = 0.f;
    #pragma unroll
    for (int w = 0; w < WAVES; ++w)
        acc += red[w * 256 + tid];
    out[(size_t)mt * 256 + tid] = acc * s;
}

extern "C" void kernel_launch(void* const* d_in, const int* in_sizes, int n_in,
                              void* d_out, int out_size, void* d_ws, size_t ws_size,
                              hipStream_t stream) {
    const float*  x      = (const float*)d_in[0];    // [IN_F, NCOL] f32
    const int8_t* Wq     = (const int8_t*)d_in[1];   // [OUT_F, IN_F] int8
    const float*  scaler = (const float*)d_in[2];    // scalar (device)
    float*        out    = (float*)d_out;            // [OUT_F, NCOL] f32

    const size_t xt_bytes = (size_t)IN_F * NCOL * sizeof(_Float16);  // 256 KB
    if (ws_size >= xt_bytes) {
        _Float16* xT = (_Float16*)d_ws;
        dim3 gcvt(IN_F / 256, NCOL);
        lq_xcvt_transpose<<<gcvt, 256, 0, stream>>>(x, xT);
        lq_wmma_gemm<true><<<OUT_F / 16, WAVES * 32, 0, stream>>>(Wq, (const void*)xT, scaler, out);
    } else {
        lq_wmma_gemm<false><<<OUT_F / 16, WAVES * 32, 0, stream>>>(Wq, (const void*)x, scaler, out);
    }
}
